// BlockSSM_14139032339192
// MI455X (gfx1250) — matrix operand order repaired and checked by reading, not compiled, not run on hardware
//
#include <hip/hip_runtime.h>
#include <hip/hip_bf16.h>

// Problem constants (from the reference)
#define NX 1024
#define NU 512
#define ND 256
#define NY 256
#define TT 64
#define BB 1024
#define KC (NX + NU + ND) // 1792 concatenated K

typedef __attribute__((ext_vector_type(16))) __bf16 v16bf;
typedef __attribute__((ext_vector_type(2)))  __bf16 v2bf;
typedef __attribute__((ext_vector_type(8)))  float  v8f;

static __device__ __forceinline__ __bf16 f2bf(float f) {
  // round-to-nearest-even fp32 -> bf16 (used only in weight packing)
  unsigned u = __builtin_bit_cast(unsigned, f);
  unsigned r = u + 0x7FFFu + ((u >> 16) & 1u);
  unsigned short h = (unsigned short)(r >> 16);
  return __builtin_bit_cast(__bf16, h);
}

static __device__ __forceinline__ v2bf pk2(float lo, float hi) {
#if __has_builtin(__builtin_amdgcn_cvt_pk_bf16_f32)
  return __builtin_amdgcn_cvt_pk_bf16_f32(lo, hi);   // hardware RNE pack
#else
  // round-half-up bf16 (ties are measure-zero), packed with one v_perm_b32:
  // out = { (hi+0x8000)[31:16], (lo+0x8000)[31:16] }
  unsigned ul = __builtin_bit_cast(unsigned, lo) + 0x8000u;
  unsigned uh = __builtin_bit_cast(unsigned, hi) + 0x8000u;
  unsigned p  = __builtin_amdgcn_perm(uh, ul, 0x07060302u);
  return __builtin_bit_cast(v2bf, p);
#endif
}

// Raw fp32 A fragment for one 16x32 WMMA A tile.
// srcl is the lane's row pointer already offset by half*16.
struct ARaw { float4 f0, f1, f2, f3; };

static __device__ __forceinline__ ARaw lda(const float* __restrict__ srcl, int kl) {
  ARaw r;
  r.f0 = *(const float4*)(srcl + kl);
  r.f1 = *(const float4*)(srcl + kl + 4);
  r.f2 = *(const float4*)(srcl + kl + 8);
  r.f3 = *(const float4*)(srcl + kl + 12);
  return r;
}

static __device__ __forceinline__ v16bf cvt_a(const ARaw& r) {
  union { v16bf v; v2bf p[8]; } av;
  av.p[0] = pk2(r.f0.x, r.f0.y); av.p[1] = pk2(r.f0.z, r.f0.w);
  av.p[2] = pk2(r.f1.x, r.f1.y); av.p[3] = pk2(r.f1.z, r.f1.w);
  av.p[4] = pk2(r.f2.x, r.f2.y); av.p[5] = pk2(r.f2.z, r.f2.w);
  av.p[6] = pk2(r.f3.x, r.f3.y); av.p[7] = pk2(r.f3.z, r.f3.w);
  return av.v;
}

#define WMMA_BF16(c, a, b) \
  __builtin_amdgcn_wmma_f32_16x16x32_bf16(false, (a), false, (b), (short)0, (c), false, false)

// ---------------------------------------------------------------------------
// Pipelined GEMM over one K-segment of NCHUNK 32-wide chunks (NCHUNK even).
// Unroll-by-2 ping-pong: consume P while loading Q, consume Q while loading P.
// No buffer rotation copies; w* pointers are pre-offset by (row, half*16).
// ---------------------------------------------------------------------------
template <int NCHUNK>
static __device__ __forceinline__ void gemm_seg(
    const float* __restrict__ srcl,
    const __bf16* __restrict__ w0, const __bf16* __restrict__ w1,
    const __bf16* __restrict__ w2, const __bf16* __restrict__ w3,
    v8f& c0, v8f& c1, v8f& c2, v8f& c3) {
  ARaw  pa  = lda(srcl, 0);
  v16bf pb0 = *(const v16bf*)(w0);
  v16bf pb1 = *(const v16bf*)(w1);
  v16bf pb2 = *(const v16bf*)(w2);
  v16bf pb3 = *(const v16bf*)(w3);

  for (int i = 0; i < NCHUNK; i += 2) {
    // Load chunk i+1 (always valid: NCHUNK even), consume chunk i.
    const int k1 = (i + 1) << 5;
    ARaw  qa  = lda(srcl, k1);
    v16bf qb0 = *(const v16bf*)(w0 + k1);
    v16bf qb1 = *(const v16bf*)(w1 + k1);
    v16bf qb2 = *(const v16bf*)(w2 + k1);
    v16bf qb3 = *(const v16bf*)(w3 + k1);
    {
      const v16bf a = cvt_a(pa);
      c0 = WMMA_BF16(c0, a, pb0);
      c1 = WMMA_BF16(c1, a, pb1);
      c2 = WMMA_BF16(c2, a, pb2);
      c3 = WMMA_BF16(c3, a, pb3);
    }
    // Load chunk i+2 (clamped to a harmless reload on the final pair),
    // consume chunk i+1.
    const int k2 = (i + 2 < NCHUNK) ? ((i + 2) << 5) : k1;
    pa  = lda(srcl, k2);
    pb0 = *(const v16bf*)(w0 + k2);
    pb1 = *(const v16bf*)(w1 + k2);
    pb2 = *(const v16bf*)(w2 + k2);
    pb3 = *(const v16bf*)(w3 + k2);
    {
      const v16bf a = cvt_a(qa);
      c0 = WMMA_BF16(c0, a, qb0);
      c1 = WMMA_BF16(c1, a, qb1);
      c2 = WMMA_BF16(c2, a, qb2);
      c3 = WMMA_BF16(c3, a, qb3);
    }
  }
}

// ---------------------------------------------------------------------------
// One recurrence step:  xout = [xprev | u | d] @ Wcat^T + b'
// Wcat (NX x KC) bf16 row-major, pre-scaled: [2Wx | 2Wu | Wd]; b' = 2bx+2bu+bd.
// Wave computes a 16(M) x 64(N) strip; three branch-free segment loops.
// ---------------------------------------------------------------------------
__global__ __launch_bounds__(256) void ssm_step_kernel(
    const float* __restrict__ xprev, const float* __restrict__ u,
    const float* __restrict__ dvec,  const __bf16* __restrict__ wcat,
    const float* __restrict__ biasp, float* __restrict__ xout) {
  const int lane = threadIdx.x & 31;
  const int wave = threadIdx.x >> 5;
  const int gw   = blockIdx.x * 8 + wave;
  const int mtile = gw >> 4;   // 64 M-tiles (B=1024 / 16)
  const int nsup  = gw & 15;   // 16 N-supertiles of 64 (NX=1024 / 64)
  const int mbase = mtile * 16;
  const int nbase = nsup * 64;
  const int half  = lane >> 4; // 0|1
  const int mr    = lane & 15; // A row in tile / B-C column in tile

  const size_t arow = (size_t)(mbase + mr);
  const float* xr = xprev + arow * NX + half * 16;
  const float* ur = u     + arow * NU + half * 16;
  const float* dr = dvec  + arow * ND + half * 16;

  const __bf16* w0 = wcat + (size_t)(nbase +  0 + mr) * KC + half * 16;
  const __bf16* w1 = wcat + (size_t)(nbase + 16 + mr) * KC + half * 16;
  const __bf16* w2 = wcat + (size_t)(nbase + 32 + mr) * KC + half * 16;
  const __bf16* w3 = wcat + (size_t)(nbase + 48 + mr) * KC + half * 16;

  v8f c0 = {}, c1 = {}, c2 = {}, c3 = {};

  gemm_seg<NX / 32>(xr, w0, w1, w2, w3, c0, c1, c2, c3);
  gemm_seg<NU / 32>(ur, w0 + NX, w1 + NX, w2 + NX, w3 + NX, c0, c1, c2, c3);
  gemm_seg<ND / 32>(dr, w0 + NX + NU, w1 + NX + NU, w2 + NX + NU, w3 + NX + NU,
                    c0, c1, c2, c3);

  const float bn0 = biasp[nbase +  0 + mr];
  const float bn1 = biasp[nbase + 16 + mr];
  const float bn2 = biasp[nbase + 32 + mr];
  const float bn3 = biasp[nbase + 48 + mr];

  // C/D layout: VGPR r -> row (r + 8*half), col = lane&15
  #pragma unroll
  for (int r = 0; r < 8; ++r) {
    float* o = xout + (size_t)(mbase + half * 8 + r) * NX + nbase + mr;
    o[0]  = c0[r] + bn0;
    o[16] = c1[r] + bn1;
    o[32] = c2[r] + bn2;
    o[48] = c3[r] + bn3;
  }
}

// ---------------------------------------------------------------------------
// Y = X @ Wy^T + by, batched over all T: M = T*B = 65536, N = 256, K = 1024.
// ---------------------------------------------------------------------------
__global__ __launch_bounds__(256) void ssm_y_kernel(
    const float* __restrict__ X, const __bf16* __restrict__ wy,
    const float* __restrict__ by, float* __restrict__ Y) {
  const int lane = threadIdx.x & 31;
  const int wave = threadIdx.x >> 5;
  const int gw   = blockIdx.x * 8 + wave;
  const int mtile = gw >> 2;  // 4096 M-tiles
  const int nsup  = gw & 3;   // 4 N-supertiles (NY=256 / 64)
  const int mbase = mtile * 16;
  const int nbase = nsup * 64;
  const int half  = lane >> 4;
  const int mr    = lane & 15;

  const float* xr = X + (size_t)(mbase + mr) * NX + half * 16;
  const __bf16* w0 = wy + (size_t)(nbase +  0 + mr) * NX + half * 16;
  const __bf16* w1 = wy + (size_t)(nbase + 16 + mr) * NX + half * 16;
  const __bf16* w2 = wy + (size_t)(nbase + 32 + mr) * NX + half * 16;
  const __bf16* w3 = wy + (size_t)(nbase + 48 + mr) * NX + half * 16;

  v8f c0 = {}, c1 = {}, c2 = {}, c3 = {};

  gemm_seg<NX / 32>(xr, w0, w1, w2, w3, c0, c1, c2, c3);

  const float bn0 = by[nbase +  0 + mr];
  const float bn1 = by[nbase + 16 + mr];
  const float bn2 = by[nbase + 32 + mr];
  const float bn3 = by[nbase + 48 + mr];

  #pragma unroll
  for (int r = 0; r < 8; ++r) {
    float* o = Y + (size_t)(mbase + half * 8 + r) * NY + nbase + mr;
    o[0]  = c0[r] + bn0;
    o[16] = c1[r] + bn1;
    o[32] = c2[r] + bn2;
    o[48] = c3[r] + bn3;
  }
}

// ---------------------------------------------------------------------------
// Weight prep: Wcat = [2Wx | 2Wu | Wd] (bf16, row-major NX x KC),
// b' = 2bx + 2bu + bd, and reg_error = 0.
// ---------------------------------------------------------------------------
__global__ void pack_wcat_kernel(const float* __restrict__ Wx, const float* __restrict__ Wu,
                                 const float* __restrict__ Wd, const float* __restrict__ bx,
                                 const float* __restrict__ bu, const float* __restrict__ bd,
                                 __bf16* __restrict__ wcat, float* __restrict__ biasp,
                                 float* __restrict__ regout) {
  const int idx = blockIdx.x * 256 + threadIdx.x;
  if (idx < NX * KC) {
    const int n = idx / KC;
    const int k = idx % KC;
    float v;
    if (k < NX)           v = 2.0f * Wx[(size_t)n * NX + k];
    else if (k < NX + NU) v = 2.0f * Wu[(size_t)n * NU + (k - NX)];
    else                  v =        Wd[(size_t)n * ND + (k - NX - NU)];
    wcat[idx] = f2bf(v);
  }
  if (idx < NX) biasp[idx] = 2.0f * bx[idx] + 2.0f * bu[idx] + bd[idx];
  if (idx == 0) *regout = 0.0f; // all Q weights are zero in the reference
}

__global__ void pack_wy_kernel(const float* __restrict__ Wy, __bf16* __restrict__ wyb) {
  const int idx = blockIdx.x * 256 + threadIdx.x;
  if (idx < NY * NX) wyb[idx] = f2bf(Wy[idx]);
}

// ---------------------------------------------------------------------------
extern "C" void kernel_launch(void* const* d_in, const int* in_sizes, int n_in,
                              void* d_out, int out_size, void* d_ws, size_t ws_size,
                              hipStream_t stream) {
  (void)in_sizes; (void)n_in; (void)out_size; (void)ws_size;
  const float* x  = (const float*)d_in[0];
  const float* U  = (const float*)d_in[1];
  const float* D  = (const float*)d_in[2];
  const float* Wx = (const float*)d_in[3];
  const float* bx = (const float*)d_in[4];
  const float* Wu = (const float*)d_in[5];
  const float* bu = (const float*)d_in[6];
  const float* Wd = (const float*)d_in[7];
  const float* bd = (const float*)d_in[8];
  const float* Wy = (const float*)d_in[9];
  const float* by = (const float*)d_in[10];

  float* out  = (float*)d_out;
  float* Xout = out;                                   // (T, B, NX) fp32
  float* Yout = out + (size_t)TT * BB * NX;            // (T, B, NY) fp32
  float* reg  = Yout + (size_t)TT * BB * NY;           // scalar

  // Workspace layout (256B aligned), ~4.2 MB total
  char* ws = (char*)d_ws;
  size_t off = 0;
  __bf16* wcat = (__bf16*)(ws + off); off += ((size_t)NX * KC * 2 + 255) & ~(size_t)255;
  __bf16* wyb  = (__bf16*)(ws + off); off += ((size_t)NY * NX * 2 + 255) & ~(size_t)255;
  float*  biasp = (float*)(ws + off);

  pack_wcat_kernel<<<(NX * KC + 255) / 256, 256, 0, stream>>>(Wx, Wu, Wd, bx, bu, bd,
                                                              wcat, biasp, reg);
  pack_wy_kernel<<<(NY * NX + 255) / 256, 256, 0, stream>>>(Wy, wyb);

  // Serial recurrence: X[t] = Wcat @ [X[t-1]|U[t]|D[t]] + b'
  // 1024 waves/step = 128 blocks of 8 waves.
  for (int t = 0; t < TT; ++t) {
    const float* xp = (t == 0) ? x : (Xout + (size_t)(t - 1) * BB * NX);
    ssm_step_kernel<<<128, 256, 0, stream>>>(xp,
                                             U + (size_t)t * BB * NU,
                                             D + (size_t)t * BB * ND,
                                             wcat, biasp,
                                             Xout + (size_t)t * BB * NX);
  }

  // Batched output projection over all T: 16384 waves = 2048 blocks.
  ssm_y_kernel<<<2048, 256, 0, stream>>>(Xout, wyb, by, Yout);
}
